// GraphTransformerEncoder_70866960384544
// MI455X (gfx1250) — compile-verified, matrix operand order checked by reading, not run on hardware
//
#include <hip/hip_runtime.h>

// ---------------- types for WMMA ----------------
typedef __bf16 bf16x16 __attribute__((ext_vector_type(16)));
typedef float  f32x8   __attribute__((ext_vector_type(8)));
typedef unsigned int uint4v __attribute__((ext_vector_type(4)));

union FragU { bf16x16 v; uint4v q[2]; };

// ---------------- small device helpers ----------------
__device__ inline unsigned short f2bf(float f) {
  unsigned int u = __float_as_uint(f);
  unsigned int r = (u + 0x7FFFu + ((u >> 16) & 1u)) >> 16;   // RNE
  return (unsigned short)r;
}

__device__ inline unsigned int fkey(float f) {               // order-preserving
  unsigned int u = __float_as_uint(f);
  return (u & 0x80000000u) ? ~u : (u | 0x80000000u);
}
__device__ inline float funkey(unsigned int k) {
  unsigned int u = (k & 0x80000000u) ? (k & 0x7FFFFFFFu) : ~k;
  return __uint_as_float(u);
}
#define KEY_NEG_INF 0x007FFFFFu  // fkey(-inf)

__device__ inline float wred_sum(float v) {
  #pragma unroll
  for (int o = 16; o > 0; o >>= 1) v += __shfl_xor(v, o, 32);
  return v;
}

// ---------------- conversion kernels ----------------
__global__ void cvt_bf16_k(const float* __restrict__ A, unsigned short* __restrict__ Ab, long n) {
  long gid = (long)blockIdx.x * blockDim.x + threadIdx.x;
  if (gid < n) Ab[gid] = f2bf(A[gid]);
}

// A [M x K] fp32 -> Ab [M x Kp] bf16, zero padded
__global__ void cvt_pad_bf16_k(const float* __restrict__ A, unsigned short* __restrict__ Ab,
                               int M, int K, int Kp) {
  long gid = (long)blockIdx.x * blockDim.x + threadIdx.x;
  if (gid >= (long)M * Kp) return;
  int r = (int)(gid / Kp), c = (int)(gid % Kp);
  Ab[gid] = (c < K) ? f2bf(A[(size_t)r * K + c]) : (unsigned short)0;
}

// W [K x Nc] fp32 -> Wt [Nc x Kp] bf16 (transposed, K padded to Kp)
__global__ void wt_cvt_k(const float* __restrict__ W, unsigned short* __restrict__ Wt,
                         int K, int Nc, int Kp) {
  long gid = (long)blockIdx.x * blockDim.x + threadIdx.x;
  if (gid >= (long)Nc * Kp) return;
  int n = (int)(gid / Kp), k = (int)(gid % Kp);
  Wt[gid] = (k < K) ? f2bf(W[(size_t)k * Nc + n]) : (unsigned short)0;
}

// ---------------- WMMA GEMM: C[MxNc] = A[MxK](bf16) * Bt[NcxK](bf16)^T + bias ----------------
// One 16x16 tile per wave; 8 waves/block cover 128 columns. Fragment loads are
// branchless B128 loads (A/B 16-bit 16x32 layout: VGPRs 0-3 and 4-7 are each 16
// contiguous bytes at row + k0 + laneHi*8 and +16).
__global__ __launch_bounds__(256)
void wmma_gemm_bf16(const unsigned short* __restrict__ A,
                    const unsigned short* __restrict__ Bt,
                    const float* __restrict__ bias,
                    float* __restrict__ Cf, unsigned short* __restrict__ Cb,
                    int M, int K, int Ncols, int relu) {
  const int lane   = threadIdx.x & 31;
  const int wave   = threadIdx.x >> 5;
  const int laneHi = lane >> 4;
  const int lhalf  = lane & 15;
  const int m0 = blockIdx.x * 16;
  const int n0 = blockIdx.y * 128 + wave * 16;
  if (n0 >= Ncols) return;

  // Clamp out-of-range A rows to row 0: garbage only feeds output rows >= M,
  // which are never stored, so loads can be unconditional (no EXEC churn).
  const int m = m0 + lhalf;
  const int n = n0 + lhalf;
  const unsigned short* arow = A + (size_t)(m < M ? m : 0) * K + (laneHi << 3);
  const unsigned short* brow = Bt + (size_t)n * K + (laneHi << 3);

  f32x8 acc = {};
  for (int k0 = 0; k0 < K; k0 += 32) {
    FragU a, b;
    a.q[0] = *(const uint4v*)(arow + k0);
    a.q[1] = *(const uint4v*)(arow + k0 + 16);
    b.q[0] = *(const uint4v*)(brow + k0);
    b.q[1] = *(const uint4v*)(brow + k0 + 16);
    acc = __builtin_amdgcn_wmma_f32_16x16x32_bf16(false, a.v, false, b.v,
                                                  (short)0, acc, false, false);
  }

  const float bv = bias ? bias[n0 + lhalf] : 0.0f;
  #pragma unroll
  for (int r = 0; r < 8; ++r) {
    const int row = m0 + r + (laneHi << 3);
    if (row < M) {
      float v = acc[r] + bv;
      if (relu) v = fmaxf(v, 0.0f);
      const size_t idx = (size_t)row * Ncols + n0 + lhalf;
      if (Cf) Cf[idx] = v;
      if (Cb) Cb[idx] = f2bf(v);
    }
  }
}

// ---------------- attention init ----------------
__global__ void attn_init_k(unsigned int* __restrict__ mkey, float* __restrict__ den,
                            float* __restrict__ agg, int N) {
  long gid = (long)blockIdx.x * blockDim.x + threadIdx.x;
  if (gid >= (long)N * 128) return;
  agg[gid] = 0.0f;
  int d = (int)(gid & 127);
  if (d < 8) {
    long i = (gid >> 7) * 8 + d;
    mkey[i] = KEY_NEG_INF;
    den[i]  = 0.0f;
  }
}

// ---------------- edge alpha + segment max ----------------
__global__ __launch_bounds__(256)
void edge_alpha_k(const float* __restrict__ q, const float* __restrict__ kbuf,
                  const int* __restrict__ src, const int* __restrict__ dst,
                  const float* __restrict__ ea, const float* __restrict__ We,
                  const float* __restrict__ be,
                  float* __restrict__ alpha, unsigned int* __restrict__ mkey, int E) {
  __shared__ float sWe[12 * 128];
  __shared__ float sbe[128];
  for (int i = threadIdx.x; i < 12 * 128; i += 256) sWe[i] = We[i];
  for (int i = threadIdx.x; i < 128; i += 256) sbe[i] = be[i];
  __syncthreads();

  long gid = (long)blockIdx.x * 256 + threadIdx.x;
  if (gid >= (long)E * 8) return;
  const int e = (int)(gid >> 3), h = (int)(gid & 7);
  const int s = src[e], d = dst[e];

  float earow[12];
  #pragma unroll
  for (int j = 0; j < 12; ++j) earow[j] = ea[(size_t)e * 12 + j];

  const float* qrow = q    + (size_t)d * 128 + h * 16;
  const float* krow = kbuf + (size_t)s * 128 + h * 16;
  float dot = 0.0f;
  #pragma unroll
  for (int c = 0; c < 16; ++c) {
    const int col = h * 16 + c;
    float ev = sbe[col];
    #pragma unroll
    for (int j = 0; j < 12; ++j) ev += earow[j] * sWe[j * 128 + col];
    dot += qrow[c] * (krow[c] + ev);
  }
  const float al = dot * 0.25f;                 // 1/sqrt(16)
  alpha[gid] = al;
  atomicMax(&mkey[(size_t)d * 8 + h], fkey(al));
}

// ---------------- exp + segment denominator ----------------
__global__ void edge_expden_k(const int* __restrict__ dst, float* __restrict__ alpha,
                              const unsigned int* __restrict__ mkey,
                              float* __restrict__ den, int E) {
  long gid = (long)blockIdx.x * blockDim.x + threadIdx.x;
  if (gid >= (long)E * 8) return;
  const int e = (int)(gid >> 3), h = (int)(gid & 7);
  const int d = dst[e];
  float m = funkey(mkey[(size_t)d * 8 + h]);
  if (!isfinite(m)) m = 0.0f;
  const float ex = expf(alpha[gid] - m);
  alpha[gid] = ex;
  atomicAdd(&den[(size_t)d * 8 + h], ex);
}

// ---------------- weighted message scatter ----------------
__global__ __launch_bounds__(256)
void edge_msg_k(const float* __restrict__ vbuf,
                const int* __restrict__ src, const int* __restrict__ dst,
                const float* __restrict__ ea, const float* __restrict__ We,
                const float* __restrict__ be,
                const float* __restrict__ alpha, const float* __restrict__ den,
                float* __restrict__ agg, int E) {
  __shared__ float sWe[12 * 128];
  __shared__ float sbe[128];
  for (int i = threadIdx.x; i < 12 * 128; i += 256) sWe[i] = We[i];
  for (int i = threadIdx.x; i < 128; i += 256) sbe[i] = be[i];
  __syncthreads();

  long gid = (long)blockIdx.x * 256 + threadIdx.x;
  if (gid >= (long)E * 128) return;
  const int e = (int)(gid >> 7), dcol = (int)(gid & 127), h = dcol >> 4;
  const int s = src[e], d = dst[e];

  const float a = alpha[(size_t)e * 8 + h] / (den[(size_t)d * 8 + h] + 1e-16f);
  float ev = sbe[dcol];
  #pragma unroll
  for (int j = 0; j < 12; ++j) ev += ea[(size_t)e * 12 + j] * sWe[j * 128 + dcol];
  const float msg = (vbuf[(size_t)s * 128 + dcol] + ev) * a;
  atomicAdd(&agg[(size_t)d * 128 + dcol], msg);
}

// ---------------- beta gate + residual + LayerNorm (one node per wave) ----------------
__global__ __launch_bounds__(256)
void gate_ln_k(const float* __restrict__ agg, const float* __restrict__ xr,
               float* __restrict__ h, const float* __restrict__ Wbeta,
               const float* __restrict__ g, const float* __restrict__ b, int N) {
  const int lane = threadIdx.x & 31, wave = threadIdx.x >> 5;
  const int node = blockIdx.x * 8 + wave;
  if (node >= N) return;
  const size_t base = (size_t)node * 128;
  const int d0 = lane * 4;

  float ag[4], xv[4], hv[4];
  #pragma unroll
  for (int i = 0; i < 4; ++i) {
    ag[i] = agg[base + d0 + i];
    xv[i] = xr[base + d0 + i];
    hv[i] = h[base + d0 + i];
  }
  float dot = 0.0f;
  #pragma unroll
  for (int i = 0; i < 4; ++i) {
    const int d = d0 + i;
    dot += ag[i] * Wbeta[d] + xv[i] * Wbeta[128 + d] + (ag[i] - xv[i]) * Wbeta[256 + d];
  }
  dot = wred_sum(dot);
  const float beta = 1.0f / (1.0f + expf(-dot));

  float y[4], s = 0.0f;
  #pragma unroll
  for (int i = 0; i < 4; ++i) { y[i] = beta * xv[i] + (1.0f - beta) * ag[i] + hv[i]; s += y[i]; }
  s = wred_sum(s);
  const float mu = s * (1.0f / 128.0f);
  float vs = 0.0f;
  #pragma unroll
  for (int i = 0; i < 4; ++i) { const float t = y[i] - mu; vs += t * t; }
  vs = wred_sum(vs);
  const float inv = rsqrtf(vs * (1.0f / 128.0f) + 1e-5f);
  #pragma unroll
  for (int i = 0; i < 4; ++i) {
    const int d = d0 + i;
    h[base + d] = (y[i] - mu) * inv * g[d] + b[d];
  }
}

// ---------------- residual + LayerNorm ----------------
__global__ __launch_bounds__(256)
void res_ln_k(const float* __restrict__ f, float* __restrict__ h,
              const float* __restrict__ g, const float* __restrict__ b, int N) {
  const int lane = threadIdx.x & 31, wave = threadIdx.x >> 5;
  const int node = blockIdx.x * 8 + wave;
  if (node >= N) return;
  const size_t base = (size_t)node * 128;
  const int d0 = lane * 4;
  float y[4], s = 0.0f;
  #pragma unroll
  for (int i = 0; i < 4; ++i) { y[i] = f[base + d0 + i] + h[base + d0 + i]; s += y[i]; }
  s = wred_sum(s);
  const float mu = s * (1.0f / 128.0f);
  float vs = 0.0f;
  #pragma unroll
  for (int i = 0; i < 4; ++i) { const float t = y[i] - mu; vs += t * t; }
  vs = wred_sum(vs);
  const float inv = rsqrtf(vs * (1.0f / 128.0f) + 1e-5f);
  #pragma unroll
  for (int i = 0; i < 4; ++i) {
    const int d = d0 + i;
    h[base + d] = (y[i] - mu) * inv * g[d] + b[d];
  }
}

// ---------------- pooling ----------------
__global__ void pool_init_k(float* __restrict__ pooled, int* __restrict__ cnt, int B) {
  long gid = (long)blockIdx.x * blockDim.x + threadIdx.x;
  if (gid < (long)B * 128) pooled[gid] = 0.0f;
  if (gid < B) cnt[gid] = 0;
}
__global__ void pool_acc_k(const float* __restrict__ h, const int* __restrict__ batch,
                           float* __restrict__ pooled, int* __restrict__ cnt, int N) {
  long gid = (long)blockIdx.x * blockDim.x + threadIdx.x;
  if (gid >= (long)N * 128) return;
  const int node = (int)(gid >> 7), d = (int)(gid & 127);
  const int b = batch[node];
  atomicAdd(&pooled[(size_t)b * 128 + d], h[gid]);
  if (d == 0) atomicAdd(&cnt[b], 1);
}
__global__ void pool_fin_k(const float* __restrict__ pooled, const int* __restrict__ cnt,
                           unsigned short* __restrict__ pooledb, int B) {
  long gid = (long)blockIdx.x * blockDim.x + threadIdx.x;
  if (gid >= (long)B * 128) return;
  const int b = (int)(gid >> 7);
  const float c = fmaxf((float)cnt[b], 1.0f);
  pooledb[gid] = f2bf(pooled[gid] / c);
}

// ---------------- host orchestration ----------------
static inline unsigned int blocks_for(long n, int t) { return (unsigned int)((n + t - 1) / t); }

extern "C" void kernel_launch(void* const* d_in, const int* in_sizes, int n_in,
                              void* d_out, int out_size, void* d_ws, size_t ws_size,
                              hipStream_t stream) {
  (void)n_in; (void)ws_size;
  const int   N  = in_sizes[3];
  const int   E  = in_sizes[2] / 2;
  const int   FA = in_sizes[0] / N;            // 78
  const int   FAp = ((FA + 31) / 32) * 32;     // 96
  const int   D = 128, DFF = 512, L = 4;
  const int   B = out_size / D;

  const float* x    = (const float*)d_in[0];
  const float* ea   = (const float*)d_in[1];
  const int*   ei   = (const int*)d_in[2];
  const int*   src  = ei;
  const int*   dst  = ei + E;
  const int*   batch = (const int*)d_in[3];
  const float* Wemb = (const float*)d_in[4];
  const float* bemb = (const float*)d_in[5];
  const float* Wout = (const float*)d_in[6];
  const float* bout = (const float*)d_in[7];
  const float* Wq = (const float*)d_in[8];   const float* bq = (const float*)d_in[9];
  const float* Wk = (const float*)d_in[10];  const float* bk = (const float*)d_in[11];
  const float* Wv = (const float*)d_in[12];  const float* bv = (const float*)d_in[13];
  const float* We = (const float*)d_in[14];  const float* be = (const float*)d_in[15];
  const float* Wskip = (const float*)d_in[16]; const float* bskip = (const float*)d_in[17];
  const float* Wbeta = (const float*)d_in[18];
  const float* g1 = (const float*)d_in[19];  const float* b1 = (const float*)d_in[20];
  const float* g2 = (const float*)d_in[21];  const float* b2 = (const float*)d_in[22];
  const float* Wf1 = (const float*)d_in[23]; const float* bf1 = (const float*)d_in[24];
  const float* Wf2 = (const float*)d_in[25]; const float* bf2 = (const float*)d_in[26];

  // workspace carving
  char* ws = (char*)d_ws;
  size_t off = 0;
  auto carve = [&](size_t bytes) -> char* {
    char* p = ws + off;
    off = (off + bytes + 255) & ~(size_t)255;
    return p;
  };
  unsigned short* xb    = (unsigned short*)carve((size_t)N * FAp * 2);
  unsigned short* hb    = (unsigned short*)carve((size_t)N * D * 2);
  float* h   = (float*)carve((size_t)N * D * 4);
  float* q   = (float*)carve((size_t)N * D * 4);
  float* k   = (float*)carve((size_t)N * D * 4);
  float* v   = (float*)carve((size_t)N * D * 4);
  float* xr  = (float*)carve((size_t)N * D * 4);
  float* alpha = (float*)carve((size_t)E * 8 * 4);
  unsigned int* mkey = (unsigned int*)carve((size_t)N * 8 * 4);
  float* den = (float*)carve((size_t)N * 8 * 4);
  float* agg = (float*)carve((size_t)N * D * 4);
  unsigned short* ff1b = (unsigned short*)carve((size_t)N * DFF * 2);
  float* ff2 = (float*)carve((size_t)N * D * 4);
  float* pooled = (float*)carve((size_t)B * D * 4);
  int*   cnt    = (int*)carve((size_t)B * 4);
  unsigned short* pooledb = (unsigned short*)carve((size_t)B * D * 2);
  unsigned short* WembT = (unsigned short*)carve((size_t)D * FAp * 2);
  unsigned short* WqT = (unsigned short*)carve((size_t)D * D * 2);
  unsigned short* WkT = (unsigned short*)carve((size_t)D * D * 2);
  unsigned short* WvT = (unsigned short*)carve((size_t)D * D * 2);
  unsigned short* WsT = (unsigned short*)carve((size_t)D * D * 2);
  unsigned short* Wf1T = (unsigned short*)carve((size_t)DFF * D * 2);
  unsigned short* Wf2T = (unsigned short*)carve((size_t)D * DFF * 2);
  unsigned short* WoutT = (unsigned short*)carve((size_t)D * D * 2);

  auto gemm = [&](const unsigned short* A, const unsigned short* Bt, const float* bias,
                  float* Cf, unsigned short* Cb, int M, int K, int Nc, int relu) {
    dim3 grid((M + 15) / 16, Nc / 128);
    wmma_gemm_bf16<<<grid, 256, 0, stream>>>(A, Bt, bias, Cf, Cb, M, K, Nc, relu);
  };

  // ---- embedding: h = x @ Wemb + bemb ----
  cvt_pad_bf16_k<<<blocks_for((long)N * FAp, 256), 256, 0, stream>>>(x, xb, N, FA, FAp);
  wt_cvt_k<<<blocks_for((long)D * FAp, 256), 256, 0, stream>>>(Wemb, WembT, FA, D, FAp);
  gemm(xb, WembT, bemb, h, nullptr, N, FAp, D, 0);

  for (int l = 0; l < L; ++l) {
    const float* Wq_l = Wq + (size_t)l * D * D;   const float* bq_l = bq + (size_t)l * D;
    const float* Wk_l = Wk + (size_t)l * D * D;   const float* bk_l = bk + (size_t)l * D;
    const float* Wv_l = Wv + (size_t)l * D * D;   const float* bv_l = bv + (size_t)l * D;
    const float* Ws_l = Wskip + (size_t)l * D * D; const float* bs_l = bskip + (size_t)l * D;
    const float* We_l = We + (size_t)l * 12 * D;  const float* be_l = be + (size_t)l * D;
    const float* Wb_l = Wbeta + (size_t)l * 3 * D;
    const float* g1_l = g1 + (size_t)l * D;       const float* b1_l = b1 + (size_t)l * D;
    const float* g2_l = g2 + (size_t)l * D;       const float* b2_l = b2 + (size_t)l * D;
    const float* Wf1_l = Wf1 + (size_t)l * D * DFF; const float* bf1_l = bf1 + (size_t)l * DFF;
    const float* Wf2_l = Wf2 + (size_t)l * DFF * D; const float* bf2_l = bf2 + (size_t)l * D;

    // h -> bf16
    cvt_bf16_k<<<blocks_for((long)N * D, 256), 256, 0, stream>>>(h, hb, (long)N * D);
    // Q, K, V, skip projections via WMMA
    wt_cvt_k<<<blocks_for((long)D * D, 256), 256, 0, stream>>>(Wq_l, WqT, D, D, D);
    wt_cvt_k<<<blocks_for((long)D * D, 256), 256, 0, stream>>>(Wk_l, WkT, D, D, D);
    wt_cvt_k<<<blocks_for((long)D * D, 256), 256, 0, stream>>>(Wv_l, WvT, D, D, D);
    wt_cvt_k<<<blocks_for((long)D * D, 256), 256, 0, stream>>>(Ws_l, WsT, D, D, D);
    gemm(hb, WqT, bq_l, q, nullptr, N, D, D, 0);
    gemm(hb, WkT, bk_l, k, nullptr, N, D, D, 0);
    gemm(hb, WvT, bv_l, v, nullptr, N, D, D, 0);
    gemm(hb, WsT, bs_l, xr, nullptr, N, D, D, 0);

    // attention over edges
    attn_init_k<<<blocks_for((long)N * D, 256), 256, 0, stream>>>(mkey, den, agg, N);
    edge_alpha_k<<<blocks_for((long)E * 8, 256), 256, 0, stream>>>(
        q, k, src, dst, ea, We_l, be_l, alpha, mkey, E);
    edge_expden_k<<<blocks_for((long)E * 8, 256), 256, 0, stream>>>(dst, alpha, mkey, den, E);
    edge_msg_k<<<blocks_for((long)E * 128, 256), 256, 0, stream>>>(
        v, src, dst, ea, We_l, be_l, alpha, den, agg, E);

    // beta gate + residual + LN1
    gate_ln_k<<<blocks_for((long)N, 8), 256, 0, stream>>>(agg, xr, h, Wb_l, g1_l, b1_l, N);

    // FFN via WMMA (relu output kept in bf16 to feed second GEMM directly)
    cvt_bf16_k<<<blocks_for((long)N * D, 256), 256, 0, stream>>>(h, hb, (long)N * D);
    wt_cvt_k<<<blocks_for((long)DFF * D, 256), 256, 0, stream>>>(Wf1_l, Wf1T, D, DFF, D);
    wt_cvt_k<<<blocks_for((long)D * DFF, 256), 256, 0, stream>>>(Wf2_l, Wf2T, DFF, D, DFF);
    gemm(hb, Wf1T, bf1_l, nullptr, ff1b, N, D, DFF, 1);
    gemm(ff1b, Wf2T, bf2_l, ff2, nullptr, N, DFF, D, 0);
    res_ln_k<<<blocks_for((long)N, 8), 256, 0, stream>>>(ff2, h, g2_l, b2_l, N);
  }

  // ---- mean pooling per graph + output projection ----
  pool_init_k<<<blocks_for((long)B * D, 256), 256, 0, stream>>>(pooled, cnt, B);
  pool_acc_k<<<blocks_for((long)N * D, 256), 256, 0, stream>>>(h, batch, pooled, cnt, N);
  pool_fin_k<<<blocks_for((long)B * D, 256), 256, 0, stream>>>(pooled, cnt, pooledb, B);
  wt_cvt_k<<<blocks_for((long)D * D, 256), 256, 0, stream>>>(Wout, WoutT, D, D, D);
  gemm(pooledb, WoutT, bout, (float*)d_out, nullptr, B, D, D, 0);
}
